// MaskedTreeAutoencoder_63376537420079
// MI455X (gfx1250) — matrix operand order repaired and checked by reading, not compiled, not run on hardware
//
#include <hip/hip_runtime.h>

#define NN    262144
#define PER   4096
#define BGR   64
#define HH    128
#define NMASK 65536
#define ST    168   // sA row stride in bf16 elements (even, 16B-aligned rows)

typedef __attribute__((ext_vector_type(16))) __bf16        v16bf;
typedef __attribute__((ext_vector_type(8)))  float         v8f;
typedef __attribute__((ext_vector_type(8)))  unsigned int  v8u;

union FragU { v8u u; v16bf b; };

__device__ __forceinline__ unsigned short f2bf(float f) {
  unsigned u = __builtin_bit_cast(unsigned, f);
  u += 0x7FFFu + ((u >> 16) & 1u);          // round-to-nearest-even
  return (unsigned short)(u >> 16);
}
__device__ __forceinline__ unsigned pack2(float a, float b) {
  return (unsigned)f2bf(a) | ((unsigned)f2bf(b) << 16);
}
__device__ __forceinline__ float bf2f(unsigned short h) {
  unsigned u = ((unsigned)h) << 16;
  return __builtin_bit_cast(float, u);
}
// sum across each 16-lane half of the wave32 (rows live per-half in C layout)
__device__ __forceinline__ float red16(float s) {
  s += __shfl_xor(s, 1, 32);
  s += __shfl_xor(s, 2, 32);
  s += __shfl_xor(s, 4, 32);
  s += __shfl_xor(s, 8, 32);
  return s;
}

// A fragment (16x32 bf16, M x K) from a row-major bf16 LDS tile.
// ISA layout: lanes 0-15 hold M=lane, K {0..7,16..23}; lanes 16-31 hold M=lane-16, K {8..15,24..31}
__device__ __forceinline__ v16bf afrag(const unsigned short* tile, int kt, int lane) {
  const unsigned* t32 = (const unsigned*)tile;
  int m   = lane & 15;
  int khi = (lane >> 4) << 3;                 // +8 K for high half
  int base = (m * ST + kt * 32 + khi) >> 1;   // dword index (all terms even)
  FragU f;
#pragma unroll
  for (int p = 0; p < 8; ++p)
    f.u[p] = t32[base + p + ((p >= 4) ? 4 : 0)];   // K pairs {2p,2p+1} (+8 for p>=4)
  return f.b;
}

// B fragment (32x16 bf16, K x N) from fragment-order-swizzled LDS:
// dword index d = ((kt*8 + nt)*8 + j)*32 + lane  -> conflict-free consecutive-lane loads
__device__ __forceinline__ v16bf bfrag(const unsigned* w, int kt, int nt, int lane) {
  int base = ((kt * 8 + nt) * 8) * 32 + lane;
  FragU f;
#pragma unroll
  for (int j = 0; j < 8; ++j) f.u[j] = w[base + j * 32];
  return f.b;
}

template <int KT>
__device__ __forceinline__ void gemm_wave(const unsigned short* aTile, const unsigned* sW,
                                          int lane, v8f acc[8]) {
#pragma unroll
  for (int kt = 0; kt < KT; ++kt) {
    v16bf a = afrag(aTile, kt, lane);
#pragma unroll
    for (int nt = 0; nt < 8; ++nt) {
      v16bf b = bfrag(sW, kt, nt, lane);
      acc[nt] = __builtin_amdgcn_wmma_f32_16x16x32_bf16(
          false, a, false, b, (short)0, acc[nt], false, false);
    }
  }
}

// ---------------- small utility kernels ----------------
__global__ void k_zero(float* f) { f[blockIdx.x * 256 + threadIdx.x] = 0.f; }

__global__ void k_setf(float* f, const int* __restrict__ idx) {
  int i = blockIdx.x * 256 + threadIdx.x;
  if (i < NMASK) f[idx[i]] = 1.0f;
}

__global__ void k_root(const float* __restrict__ h, float* __restrict__ z) {
  int i = blockIdx.x * 256 + threadIdx.x;     // 8192 = B*H
  int b = i >> 7, k = i & 127;
  z[i] = h[(size_t)(b * PER) * HH + k];
}

// Convert+swizzle all weight matrices into fragment-order bf16 (once, L2-resident afterwards)
__global__ __launch_bounds__(256) void k_prep(unsigned* __restrict__ wbuf,
    const float* __restrict__ dW1, const float* __restrict__ dW2,
    const float* __restrict__ uW1, const float* __restrict__ uW2,
    const float* __restrict__ oW1, const float* __restrict__ encW,
    const float* __restrict__ decW) {
  int b = blockIdx.x, tid = threadIdx.x;
  const float* src; int K, KT; unsigned* dst;
  if (b < 16) {
    int layer = b >> 2, w = b & 3;
    const float* base = (w == 0) ? dW1 : (w == 1) ? dW2 : (w == 2) ? uW1 : uW2;
    src = base + (size_t)layer * HH * HH; K = HH; KT = 4; dst = wbuf + b * 8192;
  } else if (b == 16) { src = oW1;  K = HH;  KT = 4; dst = wbuf + 16 * 8192; }
  else if (b == 17)   { src = encW; K = 20;  KT = 1; dst = wbuf + 17 * 8192; }
  else                { src = decW; K = 148; KT = 5; dst = wbuf + 18 * 8192; }
  for (int d = tid; d < KT * 2048; d += 256) {
    int kt = d >> 11, r = d & 2047, lane = r & 31, j = (r >> 5) & 7, nt = r >> 8;
    int Nc = nt * 16 + (lane & 15);
    int K0 = kt * 32 + ((lane >> 4) << 3) + 2 * j + ((j >= 4) ? 8 : 0);
    float f0 = (K0     < K) ? src[(size_t)K0 * HH + Nc]       : 0.f;
    float f1 = (K0 + 1 < K) ? src[(size_t)(K0 + 1) * HH + Nc] : 0.f;
    dst[d] = pack2(f0, f1);
  }
}

// ---------------- encoder input projection: h = [xm, flag, 0pad] @ Wenc + b ----------------
__global__ __launch_bounds__(128) void k_enc(
    const float* __restrict__ x, const float* __restrict__ flag,
    const unsigned* __restrict__ wenc, const float* __restrict__ eb,
    float* __restrict__ hout) {
  __shared__ unsigned sW[2048];
  __shared__ unsigned short sA[4][16 * ST];
  int tid = threadIdx.x, wave = tid >> 5, lane = tid & 31;
  int r0 = blockIdx.x * 64 + wave * 16;
  { const uint4* s = (const uint4*)wenc; uint4* d = (uint4*)sW;
    for (int i = tid; i < 512; i += 128) d[i] = s[i]; }
  for (int m = 0; m < 16; ++m) {
    int g = r0 + m;
    float fl = flag[g];
    float v = 0.f;
    if (lane < 19)       v = x[(size_t)g * 19 + lane] * (1.f - fl);
    else if (lane == 19) v = fl;
    sA[wave][m * ST + lane] = f2bf(v);
  }
  __syncthreads();
  v8f acc[8]; const v8f z8 = {0,0,0,0,0,0,0,0};
#pragma unroll
  for (int nt = 0; nt < 8; ++nt) acc[nt] = z8;
  gemm_wave<1>(sA[wave], sW, lane, acc);
  int cl = lane & 15, hi8 = (lane >> 4) << 3;
#pragma unroll
  for (int j = 0; j < 8; ++j) {
    int g = r0 + j + hi8;
    float* orow = hout + (size_t)g * HH + cl;
#pragma unroll
    for (int nt = 0; nt < 8; ++nt) orow[nt * 16] = acc[nt][j] + eb[nt * 16 + cl];
  }
}

// ---------------- decoder input projection: h = [xm, flag, z[batch], 0pad] @ Wdec + b ----------------
__global__ __launch_bounds__(128) void k_dec(
    const float* __restrict__ x, const float* __restrict__ flag,
    const float* __restrict__ zroot, const unsigned* __restrict__ wdec,
    const float* __restrict__ db, float* __restrict__ hout) {
  __shared__ unsigned sW[10240];
  __shared__ unsigned short sA[4][16 * ST];
  int tid = threadIdx.x, wave = tid >> 5, lane = tid & 31;
  int r0 = blockIdx.x * 64 + wave * 16;
  { const uint4* s = (const uint4*)wdec; uint4* d = (uint4*)sW;
    for (int i = tid; i < 2560; i += 128) d[i] = s[i]; }
  const float* zb = zroot + (size_t)(blockIdx.x >> 6) * HH;   // 64 rows/block, 4096 rows/tree
  for (int m = 0; m < 16; ++m) {
    int g = r0 + m;
    float fl = flag[g];
#pragma unroll
    for (int c5 = 0; c5 < 5; ++c5) {
      int c = lane + c5 * 32;
      float v;
      if (c < 19)       v = x[(size_t)g * 19 + c] * (1.f - fl);
      else if (c == 19) v = fl;
      else if (c < 148) v = zb[c - 20];
      else              v = 0.f;
      sA[wave][m * ST + c] = f2bf(v);
    }
  }
  __syncthreads();
  v8f acc[8]; const v8f z8 = {0,0,0,0,0,0,0,0};
#pragma unroll
  for (int nt = 0; nt < 8; ++nt) acc[nt] = z8;
  gemm_wave<5>(sA[wave], sW, lane, acc);
  int cl = lane & 15, hi8 = (lane >> 4) << 3;
#pragma unroll
  for (int j = 0; j < 8; ++j) {
    int g = r0 + j + hi8;
    float* orow = hout + (size_t)g * HH + cl;
#pragma unroll
    for (int nt = 0; nt < 8; ++nt) orow[nt * 16] = acc[nt][j] + db[nt * 16 + cl];
  }
}

// ---------------- fused GIN half-layer ----------------
// hout = LN2(ReLU( (LN1(ReLU(z@W1+b1)) @ W2 + b2) + h + dir )),  z = (1+eps)*h + tree_agg
__global__ __launch_bounds__(128) void k_gin(
    const float* __restrict__ hin, float* __restrict__ hout,
    const unsigned* __restrict__ w1s, const unsigned* __restrict__ w2s,
    const float* __restrict__ b1, const float* __restrict__ lw, const float* __restrict__ lb,
    const float* __restrict__ b2, const float* __restrict__ epsp,
    const float* __restrict__ dir, const float* __restrict__ lnw, const float* __restrict__ lnb,
    int up) {
  __shared__ unsigned sW[8192];
  __shared__ unsigned short sA[4][16 * ST];
  const int tid = threadIdx.x, wave = tid >> 5, lane = tid & 31;
  const int r0 = blockIdx.x * 64 + wave * 16;
  const float eps1 = 1.0f + *epsp;

  // stage W1 (swizzled bf16, 32KB) with b128 loads
  { const uint4* s = (const uint4*)w1s; uint4* d = (uint4*)sW;
    for (int i = tid; i < 2048; i += 128) d[i] = s[i]; }

  // build z tile in bf16 (implicit binary-tree aggregation)
  unsigned* aRow32 = (unsigned*)sA[wave];
  for (int m = 0; m < 16; ++m) {
    int g = r0 + m;
    int li = g & (PER - 1);
    int base = g - li;
    float4 hv = ((const float4*)(hin + (size_t)g * HH))[lane];
    float ax = 0.f, ay = 0.f, az = 0.f, aw = 0.f;
    if (!up) {                                   // down: gather parent
      if (li > 0) {
        float4 p = ((const float4*)(hin + (size_t)(base + ((li - 1) >> 1)) * HH))[lane];
        ax = p.x; ay = p.y; az = p.z; aw = p.w;
      }
    } else {                                     // up: sum children
      int c1 = 2 * li + 1;
      if (c1 < PER) {
        float4 p = ((const float4*)(hin + (size_t)(base + c1) * HH))[lane];
        ax += p.x; ay += p.y; az += p.z; aw += p.w;
      }
      if (c1 + 1 < PER) {
        float4 p = ((const float4*)(hin + (size_t)(base + c1 + 1) * HH))[lane];
        ax += p.x; ay += p.y; az += p.z; aw += p.w;
      }
    }
    float zx = eps1 * hv.x + ax, zy = eps1 * hv.y + ay;
    float zz = eps1 * hv.z + az, zw = eps1 * hv.w + aw;
    ((uint2*)(aRow32 + m * (ST >> 1)))[lane] = make_uint2(pack2(zx, zy), pack2(zz, zw));
  }
  __syncthreads();

  // GEMM1: z @ W1
  v8f acc[8]; const v8f z8 = {0,0,0,0,0,0,0,0};
#pragma unroll
  for (int nt = 0; nt < 8; ++nt) acc[nt] = z8;
  gemm_wave<4>(sA[wave], sW, lane, acc);

  const int cl = lane & 15, hi8 = (lane >> 4) << 3;
  float bb[8], lww[8], lbb[8];
#pragma unroll
  for (int nt = 0; nt < 8; ++nt) {
    int c = nt * 16 + cl; bb[nt] = b1[c]; lww[nt] = lw[c]; lbb[nt] = lb[c];
  }
  // ReLU + LN (rows live per 16-lane half), write t back to sA as bf16
#pragma unroll
  for (int j = 0; j < 8; ++j) {
    float s = 0.f;
#pragma unroll
    for (int nt = 0; nt < 8; ++nt) {
      float r = fmaxf(acc[nt][j] + bb[nt], 0.f); acc[nt][j] = r; s += r;
    }
    s = red16(s);
    float mean = s * (1.f / 128.f), v = 0.f;
#pragma unroll
    for (int nt = 0; nt < 8; ++nt) { float d = acc[nt][j] - mean; v += d * d; }
    v = red16(v);
    float rstd = rsqrtf(v * (1.f / 128.f) + 1e-5f);
    int M = j + hi8;
#pragma unroll
    for (int nt = 0; nt < 8; ++nt) {
      float t = (acc[nt][j] - mean) * rstd * lww[nt] + lbb[nt];
      sA[wave][M * ST + nt * 16 + cl] = f2bf(t);
    }
  }
  __syncthreads();
  // reload weight buffer with W2
  { const uint4* s = (const uint4*)w2s; uint4* d = (uint4*)sW;
    for (int i = tid; i < 2048; i += 128) d[i] = s[i]; }
  __syncthreads();

  // GEMM2: t @ W2, then residual + dir + ReLU + LN
#pragma unroll
  for (int nt = 0; nt < 8; ++nt) acc[nt] = z8;
  gemm_wave<4>(sA[wave], sW, lane, acc);

  float b2c[8], dc[8], lnwc[8], lnbc[8];
#pragma unroll
  for (int nt = 0; nt < 8; ++nt) {
    int c = nt * 16 + cl; b2c[nt] = b2[c]; dc[nt] = dir[c]; lnwc[nt] = lnw[c]; lnbc[nt] = lnb[c];
  }
#pragma unroll
  for (int j = 0; j < 8; ++j) {
    int g = r0 + j + hi8;
    const float* hr = hin + (size_t)g * HH + cl;
    float s = 0.f;
#pragma unroll
    for (int nt = 0; nt < 8; ++nt) {
      float v = acc[nt][j] + b2c[nt] + hr[nt * 16] + dc[nt];
      v = fmaxf(v, 0.f); acc[nt][j] = v; s += v;
    }
    s = red16(s);
    float mean = s * (1.f / 128.f), v2 = 0.f;
#pragma unroll
    for (int nt = 0; nt < 8; ++nt) { float d = acc[nt][j] - mean; v2 += d * d; }
    v2 = red16(v2);
    float rstd = rsqrtf(v2 * (1.f / 128.f) + 1e-5f);
    float* orow = hout + (size_t)g * HH + cl;
#pragma unroll
    for (int nt = 0; nt < 8; ++nt)
      orow[nt * 16] = (acc[nt][j] - mean) * rstd * lnwc[nt] + lnbc[nt];
  }
}

// ---------------- output head: x_hat = LN(ReLU(h@oW1+ob1)) @ oW2 + ob2 ----------------
__global__ __launch_bounds__(128) void k_out(
    const float* __restrict__ hin, const unsigned* __restrict__ w1s,
    const float* __restrict__ ob1, const float* __restrict__ olw, const float* __restrict__ olb,
    const float* __restrict__ oW2, const float* __restrict__ ob2, float* __restrict__ out) {
  __shared__ unsigned sW[8192];
  __shared__ unsigned short sA[4][16 * ST];
  const int tid = threadIdx.x, wave = tid >> 5, lane = tid & 31;
  const int r0 = blockIdx.x * 64 + wave * 16;
  { const uint4* s = (const uint4*)w1s; uint4* d = (uint4*)sW;
    for (int i = tid; i < 2048; i += 128) d[i] = s[i]; }
  unsigned* aRow32 = (unsigned*)sA[wave];
  for (int m = 0; m < 16; ++m) {
    int g = r0 + m;
    float4 hv = ((const float4*)(hin + (size_t)g * HH))[lane];
    ((uint2*)(aRow32 + m * (ST >> 1)))[lane] = make_uint2(pack2(hv.x, hv.y), pack2(hv.z, hv.w));
  }
  __syncthreads();
  v8f acc[8]; const v8f z8 = {0,0,0,0,0,0,0,0};
#pragma unroll
  for (int nt = 0; nt < 8; ++nt) acc[nt] = z8;
  gemm_wave<4>(sA[wave], sW, lane, acc);
  const int cl = lane & 15, hi8 = (lane >> 4) << 3;
  float bb[8], lww[8], lbb[8];
#pragma unroll
  for (int nt = 0; nt < 8; ++nt) {
    int c = nt * 16 + cl; bb[nt] = ob1[c]; lww[nt] = olw[c]; lbb[nt] = olb[c];
  }
#pragma unroll
  for (int j = 0; j < 8; ++j) {
    float s = 0.f;
#pragma unroll
    for (int nt = 0; nt < 8; ++nt) {
      float r = fmaxf(acc[nt][j] + bb[nt], 0.f); acc[nt][j] = r; s += r;
    }
    s = red16(s);
    float mean = s * (1.f / 128.f), v = 0.f;
#pragma unroll
    for (int nt = 0; nt < 8; ++nt) { float d = acc[nt][j] - mean; v += d * d; }
    v = red16(v);
    float rstd = rsqrtf(v * (1.f / 128.f) + 1e-5f);
    int M = j + hi8;
#pragma unroll
    for (int nt = 0; nt < 8; ++nt) {
      float t = (acc[nt][j] - mean) * rstd * lww[nt] + lbb[nt];
      sA[wave][M * ST + nt * 16 + cl] = f2bf(t);
    }
  }
  __syncthreads();
  // narrow GEMM 128 -> 19 in VALU
  for (int m = 0; m < 16; ++m) {
    int g = r0 + m;
    for (int d = lane; d < 19; d += 32) {
      float s = ob2[d];
      for (int k = 0; k < HH; ++k)
        s += bf2f(sA[wave][m * ST + k]) * oW2[(size_t)k * 19 + d];
      out[(size_t)g * 19 + d] = s;
    }
  }
}

extern "C" void kernel_launch(void* const* d_in, const int* in_sizes, int n_in,
                              void* d_out, int out_size, void* d_ws, size_t ws_size,
                              hipStream_t stream) {
  const float* x    = (const float*)d_in[0];
  const int* mask_idx = (const int*)d_in[4];
  const float* encW = (const float*)d_in[5];
  const float* encb = (const float*)d_in[6];
  const float* dW1 = (const float*)d_in[7];  const float* db1 = (const float*)d_in[8];
  const float* dlw = (const float*)d_in[9];  const float* dlb = (const float*)d_in[10];
  const float* dW2 = (const float*)d_in[11]; const float* db2 = (const float*)d_in[12];
  const float* deps = (const float*)d_in[13];
  const float* uW1 = (const float*)d_in[14]; const float* ub1 = (const float*)d_in[15];
  const float* ulw = (const float*)d_in[16]; const float* ulb = (const float*)d_in[17];
  const float* uW2 = (const float*)d_in[18]; const float* ub2 = (const float*)d_in[19];
  const float* ueps = (const float*)d_in[20];
  const float* ln1w = (const float*)d_in[21]; const float* ln1b = (const float*)d_in[22];
  const float* ln2w = (const float*)d_in[23]; const float* ln2b = (const float*)d_in[24];
  const float* dire = (const float*)d_in[25];
  const float* decW = (const float*)d_in[26]; const float* decb = (const float*)d_in[27];
  const float* oW1 = (const float*)d_in[28]; const float* ob1 = (const float*)d_in[29];
  const float* olw = (const float*)d_in[30]; const float* olb = (const float*)d_in[31];
  const float* oW2 = (const float*)d_in[32]; const float* ob2 = (const float*)d_in[33];
  float* xhat = (float*)d_out;

  float* hA   = (float*)d_ws;
  float* hB   = hA + (size_t)NN * HH;
  float* flag = hB + (size_t)NN * HH;
  float* zroot = flag + NN;
  unsigned* wbuf = (unsigned*)(zroot + BGR * HH);

  k_zero<<<NN / 256, 256, 0, stream>>>(flag);
  k_setf<<<NMASK / 256, 256, 0, stream>>>(flag, mask_idx);
  k_prep<<<19, 256, 0, stream>>>(wbuf, dW1, dW2, uW1, uW2, oW1, encW, decW);

  k_enc<<<NN / 64, 128, 0, stream>>>(x, flag, wbuf + 17 * 8192, encb, hA);
  for (int i = 0; i < 2; ++i) {   // encoder layers: hA -> hB -> hA
    k_gin<<<NN / 64, 128, 0, stream>>>(hA, hB,
        wbuf + (4 * i + 0) * 8192, wbuf + (4 * i + 1) * 8192,
        db1 + i * HH, dlw + i * HH, dlb + i * HH, db2 + i * HH, deps + i,
        dire + (i * 2 + 0) * HH, ln1w + i * HH, ln1b + i * HH, 0);
    k_gin<<<NN / 64, 128, 0, stream>>>(hB, hA,
        wbuf + (4 * i + 2) * 8192, wbuf + (4 * i + 3) * 8192,
        ub1 + i * HH, ulw + i * HH, ulb + i * HH, ub2 + i * HH, ueps + i,
        dire + (i * 2 + 1) * HH, ln2w + i * HH, ln2b + i * HH, 1);
  }
  k_root<<<BGR * HH / 256, 256, 0, stream>>>(hA, zroot);
  k_dec<<<NN / 64, 128, 0, stream>>>(x, flag, zroot, wbuf + 18 * 8192, decb, hB);
  for (int i = 2; i < 4; ++i) {   // decoder layers: hB -> hA -> hB
    k_gin<<<NN / 64, 128, 0, stream>>>(hB, hA,
        wbuf + (4 * i + 0) * 8192, wbuf + (4 * i + 1) * 8192,
        db1 + i * HH, dlw + i * HH, dlb + i * HH, db2 + i * HH, deps + i,
        dire + (i * 2 + 0) * HH, ln1w + i * HH, ln1b + i * HH, 0);
    k_gin<<<NN / 64, 128, 0, stream>>>(hA, hB,
        wbuf + (4 * i + 2) * 8192, wbuf + (4 * i + 3) * 8192,
        ub1 + i * HH, ulw + i * HH, ulb + i * HH, ub2 + i * HH, ueps + i,
        dire + (i * 2 + 1) * HH, ln2w + i * HH, ln2b + i * HH, 1);
  }
  k_out<<<NN / 64, 128, 0, stream>>>(hB, wbuf + 16 * 8192, ob1, olw, olb, oW2, ob2, xhat);
}
